// ConformerBlock_9560597201614
// MI455X (gfx1250) — matrix operand order repaired and checked by reading, not compiled, not run on hardware
//
#include <hip/hip_runtime.h>

// ---------------------------------------------------------------------------
// Conformer block for MI455X (gfx1250, wave32, WMMA).
// All big GEMMs use v_wmma_f32_16x16x32_bf16 (bf16 in, f32 accumulate).
// ---------------------------------------------------------------------------

#define Dm   512
#define Fm   2048
#define Tm   1024
#define Bm   8
#define Hm   8
#define DKm  64
#define NROW (Bm * Tm)   // 8192

typedef __attribute__((ext_vector_type(16))) __bf16 v16bf;
typedef __attribute__((ext_vector_type(8)))  float  v8f;

union AFrag { v16bf v; uint4 q[2]; };

__device__ inline unsigned short f2bf(float f) {
  union { float f; unsigned int u; } c; c.f = f;
  unsigned int u = c.u;
  u += 0x7fffu + ((u >> 16) & 1u);          // round-to-nearest-even
  return (unsigned short)(u >> 16);
}
__device__ inline float bf2f(unsigned short s) {
  union { unsigned int u; float f; } c; c.u = ((unsigned int)s) << 16;
  return c.f;
}
__device__ inline v8f zero8() {
  v8f z = {0.f, 0.f, 0.f, 0.f, 0.f, 0.f, 0.f, 0.f};
  return z;
}
__device__ inline float sigf(float x) { return 1.f / (1.f + __expf(-x)); }

__device__ inline void stv(float* p, float v)          { *p = v; }
__device__ inline void stv(unsigned short* p, float v) { *p = f2bf(v); }

// ---------------------------------------------------------------------------
// Weight transpose: in fp32 [K,N] row-major -> out bf16 [N,K] row-major (B^T)
// ---------------------------------------------------------------------------
__global__ __launch_bounds__(256) void transpose_bf16_kernel(
    const float* __restrict__ in, unsigned short* __restrict__ out,
    int Kd, int Nd) {
  __shared__ float tile[32][33];
  const int n0 = blockIdx.x * 32, k0 = blockIdx.y * 32;
  const int tx = threadIdx.x & 31, ty = threadIdx.x >> 5;  // 32 x 8
#pragma unroll
  for (int i = 0; i < 32; i += 8)
    tile[ty + i][tx] = in[(size_t)(k0 + ty + i) * Nd + (n0 + tx)];
  __syncthreads();
#pragma unroll
  for (int i = 0; i < 32; i += 8)
    out[(size_t)(n0 + ty + i) * Kd + (k0 + tx)] = f2bf(tile[tx][ty + i]);
}

// ---------------------------------------------------------------------------
// Row LayerNorm over D=512: one block (256 thr) per row, out fp32 or bf16.
// ---------------------------------------------------------------------------
template <typename OUT>
__global__ __launch_bounds__(256) void ln_kernel(
    const float* __restrict__ x, const float* __restrict__ g,
    const float* __restrict__ bta, OUT* __restrict__ out) {
  const int tid  = threadIdx.x;
  const int lane = tid & 31, wid = tid >> 5;
  const float* xp = x + (size_t)blockIdx.x * Dm;
  float v0 = xp[tid], v1 = xp[tid + 256];
  float s = v0 + v1, sq = v0 * v0 + v1 * v1;
#pragma unroll
  for (int m = 1; m < 32; m <<= 1) {
    s  += __shfl_xor(s, m);
    sq += __shfl_xor(sq, m);
  }
  __shared__ float sb[8], qb[8];
  if (lane == 0) { sb[wid] = s; qb[wid] = sq; }
  __syncthreads();
  float ts = 0.f, tq = 0.f;
#pragma unroll
  for (int i = 0; i < 8; ++i) { ts += sb[i]; tq += qb[i]; }
  const float mean = ts * (1.f / 512.f);
  const float var  = tq * (1.f / 512.f) - mean * mean;
  const float rstd = rsqrtf(var + 1e-5f);
  OUT* op = out + (size_t)blockIdx.x * Dm;
  stv(op + tid,       (v0 - mean) * rstd * g[tid]       + bta[tid]);
  stv(op + tid + 256, (v1 - mean) * rstd * g[tid + 256] + bta[tid + 256]);
}

// ---------------------------------------------------------------------------
// bf16 WMMA GEMM:  C = A[M,K] * B  (B given transposed as BT[N,K], bf16).
// Block 256 thr = 8 waves, tile 128x128, BK=32; wave tile 32x64 (2x4 frags).
// ---------------------------------------------------------------------------
#define GBM 128
#define GBN 128
#define GBK 32
#define LDT 40   // padded LDS row stride (elements) -> conflict-free frags

enum { EPI_SWISH_BF16 = 0, EPI_RES_F32 = 1, EPI_F32 = 2, EPI_QK = 3, EPI_VT = 4 };

template <int EPI>
__global__ __launch_bounds__(256) void gemm_bf16_kernel(
    const unsigned short* __restrict__ A,   // [M,K] bf16
    const unsigned short* __restrict__ BT,  // [N,K] bf16
    const float* __restrict__ bias,         // [N]
    const float* __restrict__ res,          // [M,N] fp32 (residual) or null
    float scale,
    float* __restrict__ outF,
    unsigned short* __restrict__ outB,
    int M, int N, int K) {
  __shared__ unsigned short As[GBM * LDT];
  __shared__ unsigned short Bs[GBN * LDT];
  const int tid  = threadIdx.x;
  const int lane = tid & 31, wid = tid >> 5;
  const int lh   = lane >> 4, l16 = lane & 15;     // half, lane-in-half
  const int waveM = wid >> 1, waveN = wid & 1;
  const int bm = blockIdx.x * GBM, bn = blockIdx.y * GBN;

  v8f acc[2][4];
#pragma unroll
  for (int i = 0; i < 2; ++i)
#pragma unroll
    for (int j = 0; j < 4; ++j) acc[i][j] = zero8();

  const int lrow = tid >> 1;            // 0..127
  const int lcol = (tid & 1) * 16;      // 0 or 16

  for (int k0 = 0; k0 < K; k0 += GBK) {
    const uint4* ga = (const uint4*)(A  + (size_t)(bm + lrow) * K + k0 + lcol);
    const uint4* gb = (const uint4*)(BT + (size_t)(bn + lrow) * K + k0 + lcol);
    uint4 a0 = ga[0], a1 = ga[1];
    uint4 b0 = gb[0], b1 = gb[1];
    __syncthreads();
    *(uint4*)&As[lrow * LDT + lcol]     = a0;
    *(uint4*)&As[lrow * LDT + lcol + 8] = a1;
    *(uint4*)&Bs[lrow * LDT + lcol]     = b0;
    *(uint4*)&Bs[lrow * LDT + lcol + 8] = b1;
    __syncthreads();

    AFrag af[2], bf[4];
#pragma unroll
    for (int mt = 0; mt < 2; ++mt) {
      const unsigned short* rp = &As[(waveM * 32 + mt * 16 + l16) * LDT];
      af[mt].q[0] = *(const uint4*)(rp + lh * 8);
      af[mt].q[1] = *(const uint4*)(rp + 16 + lh * 8);
    }
#pragma unroll
    for (int nt = 0; nt < 4; ++nt) {
      const unsigned short* rp = &Bs[(waveN * 64 + nt * 16 + l16) * LDT];
      bf[nt].q[0] = *(const uint4*)(rp + lh * 8);
      bf[nt].q[1] = *(const uint4*)(rp + 16 + lh * 8);
    }
#pragma unroll
    for (int mt = 0; mt < 2; ++mt)
#pragma unroll
      for (int nt = 0; nt < 4; ++nt)
        acc[mt][nt] = __builtin_amdgcn_wmma_f32_16x16x32_bf16(
            false, af[mt].v, false, bf[nt].v, (short)0, acc[mt][nt], false, false);
  }

  // epilogue (C layout: vgpr r -> m = r + 8*half, lane&15 -> n)
#pragma unroll
  for (int mt = 0; mt < 2; ++mt)
#pragma unroll
    for (int nt = 0; nt < 4; ++nt) {
      const int col = bn + waveN * 64 + nt * 16 + l16;
      const float bc = bias[col];
#pragma unroll
      for (int r = 0; r < 8; ++r) {
        const int row = bm + waveM * 32 + mt * 16 + lh * 8 + r;
        float v = acc[mt][nt][r] + bc;
        if (EPI == EPI_SWISH_BF16) {
          outB[(size_t)row * N + col] = f2bf(v * sigf(v));
        } else if (EPI == EPI_RES_F32) {
          outF[(size_t)row * N + col] = res[(size_t)row * N + col] + scale * v;
        } else if (EPI == EPI_F32) {
          outF[(size_t)row * N + col] = v;
        } else if (EPI == EPI_QK) {
          const int bb = row >> 10, t = row & (Tm - 1);
          const int h = col >> 6, dk = col & 63;
          outB[(((size_t)(bb * Hm + h)) * Tm + t) * DKm + dk] = f2bf(v);
        } else if (EPI == EPI_VT) {
          const int bb = row >> 10, t = row & (Tm - 1);
          const int h = col >> 6, dk = col & 63;
          outB[(((size_t)(bb * Hm + h)) * DKm + dk) * Tm + t] = f2bf(v);
        }
      }
    }
}

// ---------------------------------------------------------------------------
// Flash attention: grid (T/64, B*H), block 128 (4 waves, 16 queries each).
// q,k: [B,H,T,64] bf16; vT: [B,H,64,T] bf16; out: [B*T, 512] bf16.
// ---------------------------------------------------------------------------
__global__ __launch_bounds__(128) void attn_kernel(
    const unsigned short* __restrict__ q,
    const unsigned short* __restrict__ k,
    const unsigned short* __restrict__ vT,
    unsigned short* __restrict__ out) {
  __shared__ unsigned short P[4][16 * LDT];   // per-wave 16x32 prob tile (padded)
  const int lane = threadIdx.x & 31, w = threadIdx.x >> 5;
  const int lh = lane >> 4, l16 = lane & 15;
  const int bh = blockIdx.y;                  // 0..63
  const int bb = bh >> 3, h = bh & 7;
  const int q0 = blockIdx.x * 64 + w * 16;
  const unsigned short* qp = q  + (size_t)bh * Tm * DKm;
  const unsigned short* kp = k  + (size_t)bh * Tm * DKm;
  const unsigned short* vp = vT + (size_t)bh * DKm * Tm;

  AFrag aq[2];
  {
    const unsigned short* rp = qp + (size_t)(q0 + l16) * DKm;
    aq[0].q[0] = *(const uint4*)(rp + lh * 8);
    aq[0].q[1] = *(const uint4*)(rp + 16 + lh * 8);
    aq[1].q[0] = *(const uint4*)(rp + 32 + lh * 8);
    aq[1].q[1] = *(const uint4*)(rp + 48 + lh * 8);
  }
  v8f o[4];
#pragma unroll
  for (int c = 0; c < 4; ++c) o[c] = zero8();
  float mrow[8], lrow[8];
#pragma unroll
  for (int r = 0; r < 8; ++r) { mrow[r] = -1e30f; lrow[r] = 0.f; }

  for (int kb = 0; kb < Tm; kb += 32) {
    v8f s[2];
#pragma unroll
    for (int kt = 0; kt < 2; ++kt) {
      AFrag b0, b1;
      const unsigned short* rp = kp + (size_t)(kb + kt * 16 + l16) * DKm;
      b0.q[0] = *(const uint4*)(rp + lh * 8);
      b0.q[1] = *(const uint4*)(rp + 16 + lh * 8);
      b1.q[0] = *(const uint4*)(rp + 32 + lh * 8);
      b1.q[1] = *(const uint4*)(rp + 48 + lh * 8);
      v8f z = zero8();
      z = __builtin_amdgcn_wmma_f32_16x16x32_bf16(false, aq[0].v, false, b0.v,
                                                  (short)0, z, false, false);
      z = __builtin_amdgcn_wmma_f32_16x16x32_bf16(false, aq[1].v, false, b1.v,
                                                  (short)0, z, false, false);
      s[kt] = z;
    }
    // online softmax over these 32 keys (row m = r + 8*half)
#pragma unroll
    for (int r = 0; r < 8; ++r) {
      float a0 = s[0][r] * 0.125f, a1 = s[1][r] * 0.125f;
      float mx = fmaxf(a0, a1);
      mx = fmaxf(mx, __shfl_xor(mx, 1));
      mx = fmaxf(mx, __shfl_xor(mx, 2));
      mx = fmaxf(mx, __shfl_xor(mx, 4));
      mx = fmaxf(mx, __shfl_xor(mx, 8));
      const float nm = fmaxf(mrow[r], mx);
      const float corr = __expf(mrow[r] - nm);
      mrow[r] = nm;
      const float p0 = __expf(a0 - nm), p1 = __expf(a1 - nm);
      float rs = p0 + p1;
      rs += __shfl_xor(rs, 1);
      rs += __shfl_xor(rs, 2);
      rs += __shfl_xor(rs, 4);
      rs += __shfl_xor(rs, 8);
      lrow[r] = lrow[r] * corr + rs;
#pragma unroll
      for (int c = 0; c < 4; ++c) o[c][r] *= corr;
      const int m = r + 8 * lh;
      P[w][m * LDT + l16]      = f2bf(p0);
      P[w][m * LDT + 16 + l16] = f2bf(p1);
    }
    // reload P as A fragment (wave-private LDS; compiler inserts DS waits)
    AFrag pf;
    {
      const unsigned short* rp = &P[w][l16 * LDT];
      pf.q[0] = *(const uint4*)(rp + lh * 8);
      pf.q[1] = *(const uint4*)(rp + 16 + lh * 8);
    }
#pragma unroll
    for (int c = 0; c < 4; ++c) {
      AFrag vf;
      const unsigned short* vr = vp + (size_t)(c * 16 + l16) * Tm + kb;
      vf.q[0] = *(const uint4*)(vr + lh * 8);
      vf.q[1] = *(const uint4*)(vr + 16 + lh * 8);
      o[c] = __builtin_amdgcn_wmma_f32_16x16x32_bf16(false, pf.v, false, vf.v,
                                                     (short)0, o[c], false, false);
    }
  }
#pragma unroll
  for (int c = 0; c < 4; ++c)
#pragma unroll
    for (int r = 0; r < 8; ++r) {
      const int t = q0 + r + 8 * lh;
      const int col = h * DKm + c * 16 + l16;
      out[((size_t)(bb * Tm + t)) * Dm + col] = f2bf(o[c][r] / lrow[r]);
    }
}

// ---------------------------------------------------------------------------
// GLU: pw1out fp32 [8192,1024] -> glu fp32 [8192,512]
// ---------------------------------------------------------------------------
__global__ __launch_bounds__(256) void glu_kernel(
    const float* __restrict__ pw1o, float* __restrict__ glu) {
  const size_t idx = (size_t)blockIdx.x * 256 + threadIdx.x;  // < 8192*512
  const size_t row = idx >> 9;
  const int c = (int)(idx & 511);
  const float a = pw1o[row * 1024 + c];
  const float g = pw1o[row * 1024 + 512 + c];
  glu[idx] = a * sigf(g);
}

// ---------------------------------------------------------------------------
// Depthwise conv (K=31, pad 15, per batch seq) + BN + swish -> bf16
// ---------------------------------------------------------------------------
__global__ __launch_bounds__(256) void conv_bn_swish_kernel(
    const float* __restrict__ glu, const float* __restrict__ dw_w,
    const float* __restrict__ dw_b, const float* __restrict__ bn_g,
    const float* __restrict__ bn_b, const float* __restrict__ bn_m,
    const float* __restrict__ bn_v, unsigned short* __restrict__ out) {
  const size_t idx = (size_t)blockIdx.x * 256 + threadIdx.x;  // < 8192*512
  const int d = (int)(idx & 511);
  const size_t row = idx >> 9;
  const int t = (int)(row & (Tm - 1));
  const int bb = (int)(row >> 10);
  const float* gp = glu + ((size_t)bb * Tm) * Dm + d;
  float acc = dw_b[d];
#pragma unroll
  for (int j = 0; j < 31; ++j) {
    const int tt = t + j - 15;
    if (tt >= 0 && tt < Tm) acc += gp[(size_t)tt * Dm] * dw_w[d * 31 + j];
  }
  const float sc = bn_g[d] * rsqrtf(bn_v[d] + 1e-5f);
  float v = (acc - bn_m[d]) * sc + bn_b[d];
  v = v * sigf(v);
  out[idx] = f2bf(v);
}

// ---------------------------------------------------------------------------
// Host orchestration
// ---------------------------------------------------------------------------
extern "C" void kernel_launch(void* const* d_in, const int* in_sizes, int n_in,
                              void* d_out, int out_size, void* d_ws, size_t ws_size,
                              hipStream_t stream) {
  (void)in_sizes; (void)n_in; (void)out_size; (void)ws_size;
  const float* x0        = (const float*)d_in[0];
  const float* ff1_ln_g  = (const float*)d_in[1];
  const float* ff1_ln_b  = (const float*)d_in[2];
  const float* ff1_w1    = (const float*)d_in[3];
  const float* ff1_b1    = (const float*)d_in[4];
  const float* ff1_w2    = (const float*)d_in[5];
  const float* ff1_b2    = (const float*)d_in[6];
  const float* attn_ln_g = (const float*)d_in[7];
  const float* attn_ln_b = (const float*)d_in[8];
  const float* wq        = (const float*)d_in[9];
  const float* bq        = (const float*)d_in[10];
  const float* wk        = (const float*)d_in[11];
  const float* bk        = (const float*)d_in[12];
  const float* wv        = (const float*)d_in[13];
  const float* bv        = (const float*)d_in[14];
  const float* wo        = (const float*)d_in[15];
  const float* bo        = (const float*)d_in[16];
  const float* conv_ln_g = (const float*)d_in[17];
  const float* conv_ln_b = (const float*)d_in[18];
  const float* pw1_w     = (const float*)d_in[19];
  const float* pw1_b     = (const float*)d_in[20];
  const float* dw_w      = (const float*)d_in[21];
  const float* dw_b      = (const float*)d_in[22];
  const float* bn_g      = (const float*)d_in[23];
  const float* bn_b      = (const float*)d_in[24];
  const float* bn_m      = (const float*)d_in[25];
  const float* bn_v      = (const float*)d_in[26];
  const float* pw2_w     = (const float*)d_in[27];
  const float* pw2_b     = (const float*)d_in[28];
  const float* ff2_ln_g  = (const float*)d_in[29];
  const float* ff2_ln_b  = (const float*)d_in[30];
  const float* ff2_w1    = (const float*)d_in[31];
  const float* ff2_b1    = (const float*)d_in[32];
  const float* ff2_w2    = (const float*)d_in[33];
  const float* ff2_w2b   = (const float*)d_in[34];
  const float* fin_ln_g  = (const float*)d_in[35];
  const float* fin_ln_b  = (const float*)d_in[36];

  char* base = (char*)d_ws;
  size_t off = 0;
  auto alloc = [&](size_t bytes) -> void* {
    void* p = base + off;
    off = (off + bytes + 255) & ~(size_t)255;
    return p;
  };
  const size_t NRD = (size_t)NROW * Dm;
  float* xA            = (float*)alloc(NRD * 4);
  float* xB            = (float*)alloc(NRD * 4);
  unsigned short* lnb  = (unsigned short*)alloc(NRD * 2);
  unsigned short* h1   = (unsigned short*)alloc((size_t)NROW * Fm * 2);
  unsigned short* qb   = (unsigned short*)alloc(NRD * 2);
  unsigned short* kbf  = (unsigned short*)alloc(NRD * 2);
  unsigned short* vbf  = (unsigned short*)alloc(NRD * 2);
  unsigned short* atto = (unsigned short*)alloc(NRD * 2);
  float* pw1o          = (float*)alloc((size_t)NROW * 1024 * 4);
  float* glub          = (float*)alloc(NRD * 4);
  unsigned short* cvo  = (unsigned short*)alloc(NRD * 2);
  unsigned short* w1T  = (unsigned short*)alloc((size_t)Dm * Fm * 2);
  unsigned short* w2T  = (unsigned short*)alloc((size_t)Fm * Dm * 2);
  unsigned short* wqT  = (unsigned short*)alloc((size_t)Dm * Dm * 2);
  unsigned short* wkT  = (unsigned short*)alloc((size_t)Dm * Dm * 2);
  unsigned short* wvT  = (unsigned short*)alloc((size_t)Dm * Dm * 2);
  unsigned short* woT  = (unsigned short*)alloc((size_t)Dm * Dm * 2);
  unsigned short* p1T  = (unsigned short*)alloc((size_t)Dm * 1024 * 2);
  unsigned short* p2T  = (unsigned short*)alloc((size_t)Dm * Dm * 2);
  unsigned short* f1T  = (unsigned short*)alloc((size_t)Dm * Fm * 2);
  unsigned short* f2T  = (unsigned short*)alloc((size_t)Fm * Dm * 2);

  const dim3 blk256(256), blk128(128);
  auto tgrid = [](int Kd, int Nd) { return dim3(Nd / 32, Kd / 32); };

  // weight transposes -> bf16 B^T
  transpose_bf16_kernel<<<tgrid(512, 2048), blk256, 0, stream>>>(ff1_w1, w1T, 512, 2048);
  transpose_bf16_kernel<<<tgrid(2048, 512), blk256, 0, stream>>>(ff1_w2, w2T, 2048, 512);
  transpose_bf16_kernel<<<tgrid(512, 512),  blk256, 0, stream>>>(wq, wqT, 512, 512);
  transpose_bf16_kernel<<<tgrid(512, 512),  blk256, 0, stream>>>(wk, wkT, 512, 512);
  transpose_bf16_kernel<<<tgrid(512, 512),  blk256, 0, stream>>>(wv, wvT, 512, 512);
  transpose_bf16_kernel<<<tgrid(512, 512),  blk256, 0, stream>>>(wo, woT, 512, 512);
  transpose_bf16_kernel<<<tgrid(512, 1024), blk256, 0, stream>>>(pw1_w, p1T, 512, 1024);
  transpose_bf16_kernel<<<tgrid(512, 512),  blk256, 0, stream>>>(pw2_w, p2T, 512, 512);
  transpose_bf16_kernel<<<tgrid(512, 2048), blk256, 0, stream>>>(ff2_w1, f1T, 512, 2048);
  transpose_bf16_kernel<<<tgrid(2048, 512), blk256, 0, stream>>>(ff2_w2, f2T, 2048, 512);

  const dim3 gN512(NROW / GBM, 512 / GBN);
  const dim3 gN1024(NROW / GBM, 1024 / GBN);
  const dim3 gN2048(NROW / GBM, 2048 / GBN);
  const dim3 gElem((NROW * Dm) / 256);

  // ---- FFN 1: x1 = x0 + 0.5*FFN(x0) ----
  ln_kernel<unsigned short><<<NROW, blk256, 0, stream>>>(x0, ff1_ln_g, ff1_ln_b, lnb);
  gemm_bf16_kernel<EPI_SWISH_BF16><<<gN2048, blk256, 0, stream>>>(
      lnb, w1T, ff1_b1, nullptr, 0.f, nullptr, h1, NROW, Fm, Dm);
  gemm_bf16_kernel<EPI_RES_F32><<<gN512, blk256, 0, stream>>>(
      h1, w2T, ff1_b2, x0, 0.5f, xA, nullptr, NROW, Dm, Fm);

  // ---- Attention: x2 = x1 + O(LN(x1)) @ wo ----
  ln_kernel<unsigned short><<<NROW, blk256, 0, stream>>>(xA, attn_ln_g, attn_ln_b, lnb);
  gemm_bf16_kernel<EPI_QK><<<gN512, blk256, 0, stream>>>(
      lnb, wqT, bq, nullptr, 0.f, nullptr, qb, NROW, Dm, Dm);
  gemm_bf16_kernel<EPI_QK><<<gN512, blk256, 0, stream>>>(
      lnb, wkT, bk, nullptr, 0.f, nullptr, kbf, NROW, Dm, Dm);
  gemm_bf16_kernel<EPI_VT><<<gN512, blk256, 0, stream>>>(
      lnb, wvT, bv, nullptr, 0.f, nullptr, vbf, NROW, Dm, Dm);
  attn_kernel<<<dim3(Tm / 64, Bm * Hm), blk128, 0, stream>>>(qb, kbf, vbf, atto);
  gemm_bf16_kernel<EPI_RES_F32><<<gN512, blk256, 0, stream>>>(
      atto, woT, bo, xA, 1.0f, xB, nullptr, NROW, Dm, Dm);

  // ---- Conv module: x3 = x2 + pw2(swish(BN(dwconv(GLU(pw1(LN(x2))))))) ----
  ln_kernel<unsigned short><<<NROW, blk256, 0, stream>>>(xB, conv_ln_g, conv_ln_b, lnb);
  gemm_bf16_kernel<EPI_F32><<<gN1024, blk256, 0, stream>>>(
      lnb, p1T, pw1_b, nullptr, 0.f, pw1o, nullptr, NROW, 1024, Dm);
  glu_kernel<<<gElem, blk256, 0, stream>>>(pw1o, glub);
  conv_bn_swish_kernel<<<gElem, blk256, 0, stream>>>(glub, dw_w, dw_b, bn_g, bn_b,
                                                     bn_m, bn_v, cvo);
  gemm_bf16_kernel<EPI_RES_F32><<<gN512, blk256, 0, stream>>>(
      cvo, p2T, pw2_b, xB, 1.0f, xA, nullptr, NROW, Dm, Dm);

  // ---- FFN 2: x4 = x3 + 0.5*FFN(x3) ----
  ln_kernel<unsigned short><<<NROW, blk256, 0, stream>>>(xA, ff2_ln_g, ff2_ln_b, lnb);
  gemm_bf16_kernel<EPI_SWISH_BF16><<<gN2048, blk256, 0, stream>>>(
      lnb, f1T, ff2_b1, nullptr, 0.f, nullptr, h1, NROW, Fm, Dm);
  gemm_bf16_kernel<EPI_RES_F32><<<gN512, blk256, 0, stream>>>(
      h1, f2T, ff2_w2b, xA, 0.5f, xB, nullptr, NROW, Dm, Fm);

  // ---- Final LayerNorm -> fp32 output ----
  ln_kernel<float><<<NROW, blk256, 0, stream>>>(xB, fin_ln_g, fin_ln_b, (float*)d_out);
}
